// GCNClassifier_47562467836438
// MI455X (gfx1250) — matrix operand order repaired and checked by reading, not compiled, not run on hardware
//
#include <hip/hip_runtime.h>
#include <hip/hip_bf16.h>
#include <math.h>

// MI455X / gfx1250, wave32. All matmuls route through v_wmma_f32_16x16x32_f16
// (fp32 operands converted to f16 in-register, fp32 accumulate).

typedef __attribute__((ext_vector_type(16))) _Float16 v16h;
typedef __attribute__((ext_vector_type(8)))  float    v8f;

#define LL long long

// ---------------------------------------------------------------------------
// Generic batched GEMM:  C = alpha * (A x B[^T]) + bias, optional ReLU.
// One wave computes a 16x32 C tile (A fragment reused across two WMMAs);
// block = 256 threads = 8 waves = 32x128 C tile.
// Main K loop: unconditional float4 loads (addresses clamped; OOB garbage only
// reaches C rows/cols that the store mask drops). K tail handled separately.
// Batch via blockIdx.z with (z / Hdiv, z % Hdiv) two-level offsets so that
// per-head attention views (stride dk inside a row) need no repacking.
// Fragment layouts: 16-bit A 16x32 / B 32x16 use contiguous k = 16*half + e; f32 C 16x16 per CDNA5 ISA.
// ---------------------------------------------------------------------------
template <int TRANSB>
__global__ __launch_bounds__(256)
void k_gemm(const float* __restrict__ A, const float* __restrict__ Bm,
            const float* __restrict__ bias, float* __restrict__ C,
            int M, int N, int K, int lda, int ldb, int ldc,
            LL sAb, LL sBb, LL sCb, int Hdiv, LL sAh, LL sBh, LL sCh,
            float alpha, int relu)
{
  int z  = blockIdx.z;
  int zb = z / Hdiv, zh = z - zb * Hdiv;
  const float* Ab = A  + (LL)zb * sAb + (LL)zh * sAh;
  const float* Bb = Bm + (LL)zb * sBb + (LL)zh * sBh;
  float*       Cb = C  + (LL)zb * sCb + (LL)zh * sCh;

  int w    = threadIdx.x >> 5;
  int lane = threadIdx.x & 31;
  int half = lane >> 4;       // 0: lanes 0-15, 1: lanes 16-31
  int r    = lane & 15;
  int m0 = blockIdx.y * 32  + (w & 1) * 16;
  int n0 = blockIdx.x * 128 + (w >> 1) * 32;
  if (m0 >= M || n0 >= N) return;   // wave-uniform

  int row  = m0 + r;
  int rowc = row < M ? row : M - 1;            // clamp; store mask drops OOB rows
  int col0 = n0 + r, col1 = n0 + 16 + r;
  int col0c = col0 < N ? col0 : N - 1;
  int col1c = col1 < N ? col1 : N - 1;

  const float* Arow  = Ab + (LL)rowc * lda;
  const float* Brow0 = TRANSB ? (Bb + (LL)col0c * ldb) : (Bb + col0c);
  const float* Brow1 = TRANSB ? (Bb + (LL)col1c * ldb) : (Bb + col1c);

  v8f acc0 = {}, acc1 = {};
  int K32 = K & ~31;

  for (int k0 = 0; k0 < K32; k0 += 32) {
    // ---- A 16x32 fragment: element e -> k0+half*16+e
    const float4* pa0 = (const float4*)(Arow + k0 + half * 16);
    const float4* pa1 = (const float4*)(Arow + k0 + half * 16 + 8);
    float4 aA = pa0[0], aB = pa0[1], aC = pa1[0], aD = pa1[1];
    v16h a;
    a[0]  = (_Float16)aA.x; a[1]  = (_Float16)aA.y; a[2]  = (_Float16)aA.z; a[3]  = (_Float16)aA.w;
    a[4]  = (_Float16)aB.x; a[5]  = (_Float16)aB.y; a[6]  = (_Float16)aB.z; a[7]  = (_Float16)aB.w;
    a[8]  = (_Float16)aC.x; a[9]  = (_Float16)aC.y; a[10] = (_Float16)aC.z; a[11] = (_Float16)aC.w;
    a[12] = (_Float16)aD.x; a[13] = (_Float16)aD.y; a[14] = (_Float16)aD.z; a[15] = (_Float16)aD.w;

    // ---- B 32x16 fragments: lane = column, element e -> K = k0+half*16+e
    v16h b0, b1;
    if (TRANSB) {
      const float4* pb0 = (const float4*)(Brow0 + k0 + half * 16);
      const float4* pb1 = (const float4*)(Brow1 + k0 + half * 16);
      float4 u0 = pb0[0], u1 = pb0[1], u2 = pb0[2], u3 = pb0[3];
      float4 v0 = pb1[0], v1 = pb1[1], v2 = pb1[2], v3 = pb1[3];
      b0[0]  = (_Float16)u0.x; b0[1]  = (_Float16)u0.y; b0[2]  = (_Float16)u0.z; b0[3]  = (_Float16)u0.w;
      b0[4]  = (_Float16)u1.x; b0[5]  = (_Float16)u1.y; b0[6]  = (_Float16)u1.z; b0[7]  = (_Float16)u1.w;
      b0[8]  = (_Float16)u2.x; b0[9]  = (_Float16)u2.y; b0[10] = (_Float16)u2.z; b0[11] = (_Float16)u2.w;
      b0[12] = (_Float16)u3.x; b0[13] = (_Float16)u3.y; b0[14] = (_Float16)u3.z; b0[15] = (_Float16)u3.w;
      b1[0]  = (_Float16)v0.x; b1[1]  = (_Float16)v0.y; b1[2]  = (_Float16)v0.z; b1[3]  = (_Float16)v0.w;
      b1[4]  = (_Float16)v1.x; b1[5]  = (_Float16)v1.y; b1[6]  = (_Float16)v1.z; b1[7]  = (_Float16)v1.w;
      b1[8]  = (_Float16)v2.x; b1[9]  = (_Float16)v2.y; b1[10] = (_Float16)v2.z; b1[11] = (_Float16)v2.w;
      b1[12] = (_Float16)v3.x; b1[13] = (_Float16)v3.y; b1[14] = (_Float16)v3.z; b1[15] = (_Float16)v3.w;
    } else {
#pragma unroll
      for (int e = 0; e < 16; ++e) {
        int kk = k0 + half * 16 + e;
        b0[e] = (_Float16)Brow0[(LL)kk * ldb];
        b1[e] = (_Float16)Brow1[(LL)kk * ldb];
      }
    }
    __builtin_prefetch(Arow + k0 + 128, 0, 0);  // global_prefetch_b8
    acc0 = __builtin_amdgcn_wmma_f32_16x16x32_f16(false, a, false, b0,
                                                  (short)0, acc0, false, false);
    acc1 = __builtin_amdgcn_wmma_f32_16x16x32_f16(false, a, false, b1,
                                                  (short)0, acc1, false, false);
  }

  if (K32 < K) {  // ragged K tail (e.g. K=360): clamped loads + select-zero
    v16h a, b0, b1;
#pragma unroll
    for (int e = 0; e < 8; ++e) {
      int k1 = K32 + half * 16 + e;
      int k2 = k1 + 8;
      float x1 = Arow[k1 < K ? k1 : K - 1];
      float x2 = Arow[k2 < K ? k2 : K - 1];
      a[e]     = (_Float16)(k1 < K ? x1 : 0.f);
      a[e + 8] = (_Float16)(k2 < K ? x2 : 0.f);
    }
#pragma unroll
    for (int e = 0; e < 16; ++e) {
      int kk = K32 + half * 16 + e;
      int kc = kk < K ? kk : K - 1;
      float u0 = TRANSB ? Brow0[kc] : Brow0[(LL)kc * ldb];
      float u1 = TRANSB ? Brow1[kc] : Brow1[(LL)kc * ldb];
      b0[e] = (_Float16)(kk < K ? u0 : 0.f);
      b1[e] = (_Float16)(kk < K ? u1 : 0.f);
    }
    acc0 = __builtin_amdgcn_wmma_f32_16x16x32_f16(false, a, false, b0,
                                                  (short)0, acc0, false, false);
    acc1 = __builtin_amdgcn_wmma_f32_16x16x32_f16(false, a, false, b1,
                                                  (short)0, acc1, false, false);
  }

  // ---- C 16x16 f32: VGPR v -> M = m0 + half*8 + v, N = n0(+16) + (lane&15)
  float bi0 = bias ? bias[col0c] : 0.f;
  float bi1 = bias ? bias[col1c] : 0.f;
#pragma unroll
  for (int v = 0; v < 8; ++v) {
    int mm = m0 + half * 8 + v;
    if (mm < M) {
      if (col0 < N) {
        float val = acc0[v] * alpha + bi0;
        if (relu) val = fmaxf(val, 0.f);
        Cb[(LL)mm * ldc + col0] = val;
      }
      if (col1 < N) {
        float val = acc1[v] * alpha + bi1;
        if (relu) val = fmaxf(val, 0.f);
        Cb[(LL)mm * ldc + col1] = val;
      }
    }
  }
}

// ---------------------------------------------------------------------------
// Elementwise / reduction helpers
// ---------------------------------------------------------------------------
__device__ inline float sigf(float x) { return 1.f / (1.f + expf(-x)); }

__global__ void k_zero(float* p, LL n)
{ LL i = (LL)blockIdx.x * 256 + threadIdx.x; if (i < n) p[i] = 0.f; }

__global__ void k_embed(const int* __restrict__ tok, const int* __restrict__ pos,
                        const int* __restrict__ post,
                        const float* __restrict__ emb, const float* __restrict__ pemb,
                        const float* __restrict__ qemb, float* __restrict__ x, int n)
{
  int i = blockIdx.x * 256 + threadIdx.x;
  if (i >= n) return;
  int d = i % 360, bs = i / 360;
  float v;
  if (d < 300)      v = emb [(LL)tok [bs] * 300 + d];
  else if (d < 330) v = pemb[(LL)pos [bs] * 30 + (d - 300)];
  else              v = qemb[(LL)post[bs] * 30 + (d - 330)];
  x[i] = v;
}

__global__ void k_lstm_step(const float* __restrict__ pre, const float* __restrict__ g,
                            float* h, float* c, float* rnn, int t, int coloff)
{
  int i = blockIdx.x * 256 + threadIdx.x;
  if (i >= 32 * 512) return;
  int b = i >> 9, j = i & 511;
  const float* pr = pre + ((LL)b * 256 + t) * 2048;
  const float* gb = g + (LL)b * 2048;
  float gi = pr[j]        + gb[j];
  float gf = pr[512 + j]  + gb[512 + j];
  float gg = pr[1024 + j] + gb[1024 + j];
  float go = pr[1536 + j] + gb[1536 + j];
  float cc = sigf(gf) * c[i] + sigf(gi) * tanhf(gg);
  float hh = sigf(go) * tanhf(cc);
  c[i] = cc; h[i] = hh;
  rnn[((LL)b * 256 + t) * 1024 + coloff + j] = hh;
}

__global__ void k_eq0(const float* a, float* o, int n)
{ int i = blockIdx.x * 256 + threadIdx.x; if (i < n) o[i] = (a[i] == 0.f) ? 1.f : 0.f; }

// Row softmax over 256 columns; optional score-mask (smask!=null) with
// -1e9 masking and keep = 1 - mask[row][0]. One block per row.
__global__ __launch_bounds__(256)
void k_softmax_row(float* sc, const float* smask, int HS)
{
  int rid = blockIdx.x;
  int j = threadIdx.x;
  float* row = sc + (LL)rid * 256;
  float keep = 1.f, v;
  if (smask) {
    int b = rid / HS;
    int i = rid & 255;
    const float* sm = smask + ((LL)b * 256 + i) * 256;
    v = (sm[j] > 0.f) ? -1e9f : row[j];
    keep = (sm[0] > 0.f) ? 0.f : 1.f;
  } else v = row[j];
  __shared__ float red[256];
  red[j] = v; __syncthreads();
  for (int s = 128; s > 0; s >>= 1) { if (j < s) red[j] = fmaxf(red[j], red[j + s]); __syncthreads(); }
  float mx = red[0]; __syncthreads();
  float e = expf(v - mx);
  red[j] = e; __syncthreads();
  for (int s = 128; s > 0; s >>= 1) { if (j < s) red[j] += red[j + s]; __syncthreads(); }
  row[j] = e / red[0] * keep;
}

__global__ void k_headsum(const float* sc, float* out, int n)  // n = B*S*S
{
  int i = blockIdx.x * 256 + threadIdx.x;
  if (i >= n) return;
  int b = i / 65536, rem = i - b * 65536;
  const float* base = sc + (LL)b * 8 * 65536 + rem;
  float s = 0.f;
  for (int h = 0; h < 8; ++h) s += base[(LL)h * 65536];
  out[i] = s;
}

// k-th largest of n nonnegative values via 32-step bisection; one block/batch.
__global__ __launch_bounds__(256)
void k_topk(const float* a, float* thr, int n, int k)
{
  int b = blockIdx.x, tid = threadIdx.x;
  const float* ab = a + (LL)b * n;
  __shared__ float redf[256];
  __shared__ int   redi[256];
  float mx = 0.f;
  for (int i = tid; i < n; i += 256) mx = fmaxf(mx, ab[i]);
  redf[tid] = mx; __syncthreads();
  for (int s = 128; s > 0; s >>= 1) { if (tid < s) redf[tid] = fmaxf(redf[tid], redf[tid + s]); __syncthreads(); }
  float lo = 0.f, hi = redf[0];
  __syncthreads();
  for (int it = 0; it < 32; ++it) {
    float mid = 0.5f * (lo + hi);
    int c = 0;
    for (int i = tid; i < n; i += 256) c += (ab[i] >= mid) ? 1 : 0;
    redi[tid] = c; __syncthreads();
    for (int s = 128; s > 0; s >>= 1) { if (tid < s) redi[tid] += redi[tid + s]; __syncthreads(); }
    int cnt = redi[0]; __syncthreads();
    if (cnt >= k) lo = mid; else hi = mid;
  }
  if (tid == 0) thr[b] = lo;
}

__global__ void k_select_adj(const float* a, const float* thr, float* out, int n)
{
  int i = blockIdx.x * 256 + threadIdx.x;
  if (i >= n) return;
  int b = i / 65536, rem = i - b * 65536, r = rem >> 8, c = rem & 255;
  float t = thr[b];
  const float* ab = a + (LL)b * 65536;
  float bij = (ab[r * 256 + c] >= t) ? 1.f : 0.f;
  float bji = (ab[c * 256 + r] >= t) ? 1.f : 0.f;
  float sym = (r == c) ? 1.f : (bij + bji);
  out[i] = sym * bij;
}

__global__ void k_rowsum(const float* adj, float* denom, int n)  // n = B*S
{
  int i = blockIdx.x * 256 + threadIdx.x;
  if (i >= n) return;
  const float* row = adj + (LL)i * 256;
  float s = 1.f;
  for (int j = 0; j < 256; ++j) s += row[j];
  denom[i] = s;
}

__global__ void k_gcn_combine(const float* t2, const float* t3, const float* denom,
                              float* h, int n)
{
  int i = blockIdx.x * 256 + threadIdx.x;
  if (i >= n) return;
  int r = i >> 9;
  h[i] = fmaxf(t2[i] / denom[r], 0.f) + t3[i];
}

__global__ void k_hcom(const float* hw, const float* a, const float* b, float* o, int n)
{
  int i = blockIdx.x * 256 + threadIdx.x;
  if (i < n) o[i] = (hw[0] * a[i] + hw[1] * b[i]) * 0.5f;
}

__global__ void k_copy2d(const float* src, float* dst, int rows, int cols, int lds, int ldd)
{
  LL i = (LL)blockIdx.x * 256 + threadIdx.x;
  LL tot = (LL)rows * cols;
  if (i >= tot) return;
  int r = (int)(i / cols), c = (int)(i - (LL)r * cols);
  dst[(LL)r * ldd + c] = src[(LL)r * lds + c];
}

__global__ void k_pool(const float* h, const float* mask, float* outs, int colOff)
{
  int i = blockIdx.x * 256 + threadIdx.x;
  if (i >= 32 * 512) return;
  int b = i >> 9, d = i & 511;
  const float* mrow = mask + b * 256;
  float s = 0.f, wn = 0.f;
  for (int t = 0; t < 256; ++t) { float m = mrow[t]; s += h[((LL)b * 256 + t) * 512 + d] * m; wn += m; }
  outs[(LL)b * 4608 + colOff + d] = s / wn;
}

// ---------------------------------------------------------------------------
// Host orchestration
// ---------------------------------------------------------------------------
static void gemm(hipStream_t st, const float* A, const float* B, const float* bias,
                 float* C, int M, int N, int K, int lda, int ldb, int ldc, int transB,
                 int batch = 1, LL sAb = 0, LL sBb = 0, LL sCb = 0,
                 int Hdiv = 1, LL sAh = 0, LL sBh = 0, LL sCh = 0,
                 float alpha = 1.f, int relu = 0)
{
  dim3 g((unsigned)((N + 127) / 128), (unsigned)((M + 31) / 32), (unsigned)batch);
  if (transB)
    k_gemm<1><<<g, 256, 0, st>>>(A, B, bias, C, M, N, K, lda, ldb, ldc,
                                 sAb, sBb, sCb, Hdiv, sAh, sBh, sCh, alpha, relu);
  else
    k_gemm<0><<<g, 256, 0, st>>>(A, B, bias, C, M, N, K, lda, ldb, ldc,
                                 sAb, sBb, sCb, Hdiv, sAh, sBh, sCh, alpha, relu);
}

static inline unsigned blocks(LL n) { return (unsigned)((n + 255) / 256); }

// Input leaf indices: recursive dict insertion order of setup_inputs()
enum {
  IN_TOK = 0, IN_POS, IN_POST, IN_MASK, IN_ADJ,
  P_EMB, P_POSEMB, P_POSTEMB,
  P_WIHF, P_WHHF, P_BF, P_WIHB, P_WHHB, P_BB,
  P_A0QW, P_A0QB, P_A0KW, P_A0KB,
  P_G1W0W, P_G1W0B, P_G1W1W, P_G1W1B, P_G1W2W, P_G1W2B,
  P_G2W0W, P_G2W0B, P_G2W1W, P_G2W1B, P_G2W2W, P_G2W2B,
  P_G2A0QW, P_G2A0QB, P_G2A0KW, P_G2A0KB,
  P_G2A1QW, P_G2A1QB, P_G2A1KW, P_G2A1KB,
  P_GCW0W, P_GCW0B, P_GCW1W, P_GCW1B, P_GCW2W, P_GCW2B,
  P_GCA0QW, P_GCA0QB, P_GCA0KW, P_GCA0KB,
  P_GCA1QW, P_GCA1QB, P_GCA1KW, P_GCA1KB,
  P_HW, P_LINW, P_LINB,
  P_AFF1, P_AFF2, P_AFF3,
  P_CLSW, P_CLSB
};

extern "C" void kernel_launch(void* const* d_in, const int* in_sizes, int n_in,
                              void* d_out, int out_size, void* d_ws, size_t ws_size,
                              hipStream_t stream)
{
  (void)in_sizes; (void)n_in; (void)out_size; (void)ws_size;
  auto F = [&](int i) -> const float* { return (const float*)d_in[i]; };
  auto I = [&](int i) -> const int*   { return (const int*)d_in[i]; };

  // ---- workspace bump allocation (fp32) ----
  float* wsp = (float*)d_ws;
  size_t off = 0;
  auto alloc = [&](size_t n) { float* p = wsp + off; off += n; return p; };

  float* x       = alloc((size_t)32 * 256 * 360);
  float* pre_f   = alloc((size_t)8192 * 2048);
  float* pre_b   = alloc((size_t)8192 * 2048);
  float* rnnBuf  = alloc((size_t)8192 * 1024);
  float* smask   = alloc((size_t)32 * 256 * 256);
  float* att_adj = alloc((size_t)32 * 256 * 256);
  float* adj_tmp = alloc((size_t)32 * 256 * 256);
  float* T2      = alloc((size_t)8192 * 2048);   // scores (B,H,S,S) == (8192,2048)
  float* T3      = alloc((size_t)32 * 256 * 256);
  float* denom   = alloc(8192);
  float* thr     = alloc(32);
  float* hF      = alloc(16384);
  float* cF      = alloc(16384);
  float* hB      = alloc(16384);
  float* cB      = alloc(16384);
  float* gbuf    = alloc((size_t)32 * 2048);
  float* sy_in   = alloc((size_t)8192 * 2048);
  float* se_in   = alloc((size_t)8192 * 2048);
  float* csy_in  = alloc((size_t)8192 * 2048);
  float* cse_in  = alloc((size_t)8192 * 2048);
  float* h_sy    = alloc((size_t)8192 * 512);
  float* h_se    = alloc((size_t)8192 * 512);
  float* h_csy   = alloc((size_t)8192 * 512);
  float* h_cse   = alloc((size_t)8192 * 512);
  float* h_com   = alloc((size_t)8192 * 512);
  float* h_sy1   = alloc((size_t)8192 * 512);
  float* h_se1   = alloc((size_t)8192 * 512);
  float* h_com1  = alloc((size_t)8192 * 512);
  float* h_sy2   = alloc((size_t)8192 * 512);
  float* h_se2   = alloc((size_t)8192 * 512);
  float* h_com2  = alloc((size_t)8192 * 512);
  float* outs    = alloc((size_t)32 * 4608);
  float* fin     = alloc((size_t)32 * 512);
  float* T0 = pre_f;   // q / t2 scratch after LSTM is done
  float* T1 = pre_b;   // k / t3 scratch after LSTM is done

  const int nSS = 32 * 256 * 256;

  // ---- 1. embeddings ----
  {
    int n = 32 * 256 * 360;
    k_embed<<<blocks(n), 256, 0, stream>>>(I(IN_TOK), I(IN_POS), I(IN_POST),
                                           F(P_EMB), F(P_POSEMB), F(P_POSTEMB), x, n);
  }

  // ---- 2. LSTM input projections ----
  gemm(stream, x, F(P_WIHF), F(P_BF), pre_f, 8192, 2048, 360, 360, 360, 2048, 1);
  gemm(stream, x, F(P_WIHB), F(P_BB), pre_b, 8192, 2048, 360, 360, 360, 2048, 1);

  // ---- 3. LSTM scans (sequential recurrent WMMA GEMMs) ----
  k_zero<<<blocks(4 * 16384), 256, 0, stream>>>(hF, 4 * 16384);  // hF,cF,hB,cB contiguous
  for (int t = 0; t < 256; ++t) {
    gemm(stream, hF, F(P_WHHF), nullptr, gbuf, 32, 2048, 512, 512, 512, 2048, 1);
    k_lstm_step<<<64, 256, 0, stream>>>(pre_f, gbuf, hF, cF, rnnBuf, t, 0);
  }
  for (int t = 255; t >= 0; --t) {
    gemm(stream, hB, F(P_WHHB), nullptr, gbuf, 32, 2048, 512, 512, 512, 2048, 1);
    k_lstm_step<<<64, 256, 0, stream>>>(pre_b, gbuf, hB, cB, rnnBuf, t, 512);
  }

  // ---- 4. score mask: (rnn @ rnn^T == 0) ----
  gemm(stream, rnnBuf, rnnBuf, nullptr, T2, 256, 256, 1024, 1024, 1024, 256, 1,
       32, (LL)256 * 1024, (LL)256 * 1024, (LL)256 * 256);
  k_eq0<<<blocks(nSS), 256, 0, stream>>>(T2, smask, nSS);

  // ---- helpers ----
  auto semantic_adj = [&](const float* xP, int ldx, int dim,
                          const float* qw, const float* qb,
                          const float* kw, const float* kb, float* adjOut) {
    int dk = dim / 8;
    gemm(stream, xP, qw, qb, T0, 8192, dim, dim, ldx, dim, dim, 1);
    gemm(stream, xP, kw, kb, T1, 8192, dim, dim, ldx, dim, dim, 1);
    gemm(stream, T0, T1, nullptr, T2, 256, 256, dk, dim, dim, 256, 1,
         256, (LL)256 * dim, (LL)256 * dim, (LL)8 * 65536,
         8, dk, dk, 65536, 1.f / sqrtf((float)dk), 0);
    k_softmax_row<<<32 * 8 * 256, 256, 0, stream>>>(T2, smask, 8 * 256);
    k_headsum<<<blocks(nSS), 256, 0, stream>>>(T2, T3, nSS);
    k_topk<<<32, 256, 0, stream>>>(T3, thr, 65536, 512);
    k_select_adj<<<blocks(nSS), 256, 0, stream>>>(T3, thr, adjOut, nSS);
  };

  auto gcn_layer = [&](const float* adjP, const float* xP, int ldx, int dim,
                       const float* Ww, const float* Wb, float* hOut) {
    k_rowsum<<<blocks(8192), 256, 0, stream>>>(adjP, denom, 8192);
    gemm(stream, adjP, xP, nullptr, T2, 256, dim, 256, 256, ldx, dim, 0,
         32, (LL)256 * 256, (LL)256 * ldx, (LL)256 * dim);              // Ax
    gemm(stream, T2, Ww, Wb, T0, 8192, 512, dim, dim, dim, 512, 1);     // (Ax)W + b
    gemm(stream, xP, Ww, Wb, T1, 8192, 512, dim, ldx, dim, 512, 1);     // xW + b
    k_gcn_combine<<<blocks((LL)8192 * 512), 256, 0, stream>>>(T0, T1, denom, hOut,
                                                              8192 * 512);
  };

  auto cross = [&](const float* aP, const float* aff, const float* bP, float* outP) {
    gemm(stream, aP, aff, nullptr, T0, 8192, 512, 512, 512, 512, 512, 0); // a@aff
    gemm(stream, T0, bP, nullptr, T2, 256, 256, 512, 512, 512, 256, 1,
         32, (LL)256 * 512, (LL)256 * 512, (LL)256 * 256);                 // t@bm^T
    k_softmax_row<<<32 * 256, 256, 0, stream>>>(T2, nullptr, 1);
    gemm(stream, T2, bP, nullptr, outP, 256, 512, 256, 256, 512, 512, 0,
         32, (LL)256 * 256, (LL)256 * 512, (LL)256 * 512);                 // softmax@bm
  };

  // ---- 5. attn0 -> att_adj ----
  semantic_adj(rnnBuf, 1024, 1024, F(P_A0QW), F(P_A0QB), F(P_A0KW), F(P_A0KB), att_adj);

  const float* adjIn = F(IN_ADJ);

  // ---- 6. GCN stack ----
  for (int i = 0; i < 3; ++i) {
    int dim = 1024 + 512 * i;
    const float* g1w = F(P_G1W0W + 2 * i); const float* g1b = F(P_G1W0B + 2 * i);
    const float* g2w = F(P_G2W0W + 2 * i); const float* g2b = F(P_G2W0B + 2 * i);
    const float* gcw = F(P_GCW0W + 2 * i); const float* gcb = F(P_GCW0B + 2 * i);

    if (i == 0) {
      gcn_layer(adjIn,   rnnBuf, 1024, 1024, g1w, g1b, h_sy);
      gcn_layer(att_adj, rnnBuf, 1024, 1024, g2w, g2b, h_se);
      gcn_layer(adjIn,   rnnBuf, 1024, 1024, gcw, gcb, h_csy);
      gcn_layer(att_adj, rnnBuf, 1024, 1024, gcw, gcb, h_cse);
      LL nR = (LL)8192 * 1024, nH = (LL)8192 * 512;
      k_copy2d<<<blocks(nR), 256, 0, stream>>>(rnnBuf, sy_in,  8192, 1024, 1024, 2048);
      k_copy2d<<<blocks(nR), 256, 0, stream>>>(rnnBuf, se_in,  8192, 1024, 1024, 2048);
      k_copy2d<<<blocks(nR), 256, 0, stream>>>(rnnBuf, csy_in, 8192, 1024, 1024, 2048);
      k_copy2d<<<blocks(nR), 256, 0, stream>>>(rnnBuf, cse_in, 8192, 1024, 1024, 2048);
      k_copy2d<<<blocks(nH), 256, 0, stream>>>(h_sy,  sy_in  + 1024, 8192, 512, 512, 2048);
      k_copy2d<<<blocks(nH), 256, 0, stream>>>(h_se,  se_in  + 1024, 8192, 512, 512, 2048);
      k_copy2d<<<blocks(nH), 256, 0, stream>>>(h_csy, csy_in + 1024, 8192, 512, 512, 2048);
      k_copy2d<<<blocks(nH), 256, 0, stream>>>(h_cse, cse_in + 1024, 8192, 512, 512, 2048);
    } else {
      int ab = (i == 1) ? 0 : 4;  // attn param block offset within g2/gc
      gcn_layer(adjIn, sy_in, 2048, dim, g1w, g1b, h_sy);
      semantic_adj(se_in, 2048, dim, F(P_G2A0QW + ab), F(P_G2A0QB + ab),
                   F(P_G2A0KW + ab), F(P_G2A0KB + ab), adj_tmp);
      gcn_layer(adj_tmp, se_in, 2048, dim, g2w, g2b, h_se);
      gcn_layer(adjIn, csy_in, 2048, dim, gcw, gcb, h_csy);
      semantic_adj(cse_in, 2048, dim, F(P_GCA0QW + ab), F(P_GCA0QB + ab),
                   F(P_GCA0KW + ab), F(P_GCA0KB + ab), adj_tmp);
      gcn_layer(adj_tmp, cse_in, 2048, dim, gcw, gcb, h_cse);
      if (i < 2) {  // appended columns beyond layer 2 are never consumed
        LL nH = (LL)8192 * 512;
        k_copy2d<<<blocks(nH), 256, 0, stream>>>(h_sy,  sy_in  + dim, 8192, 512, 512, 2048);
        k_copy2d<<<blocks(nH), 256, 0, stream>>>(h_se,  se_in  + dim, 8192, 512, 512, 2048);
        k_copy2d<<<blocks(nH), 256, 0, stream>>>(h_csy, csy_in + dim, 8192, 512, 512, 2048);
        k_copy2d<<<blocks(nH), 256, 0, stream>>>(h_cse, cse_in + dim, 8192, 512, 512, 2048);
      }
    }

    k_hcom<<<blocks((LL)8192 * 512), 256, 0, stream>>>(F(P_HW), h_csy, h_cse, h_com,
                                                       8192 * 512);
    cross(h_sy,  F(P_AFF1), h_se,  h_sy1);
    cross(h_se,  F(P_AFF3), h_sy,  h_se1);
    cross(h_com, F(P_AFF3), h_sy,  h_com1);
    cross(h_sy1, F(P_AFF2), h_com, h_sy2);
    cross(h_se1, F(P_AFF2), h_com, h_se2);
    cross(h_com1, F(P_AFF1), h_se, h_com2);

    k_pool<<<64, 256, 0, stream>>>(h_sy2,  F(IN_MASK), outs, (3 * i + 0) * 512);
    k_pool<<<64, 256, 0, stream>>>(h_se2,  F(IN_MASK), outs, (3 * i + 1) * 512);
    k_pool<<<64, 256, 0, stream>>>(h_com2, F(IN_MASK), outs, (3 * i + 2) * 512);
  }

  // ---- 7. classifier head ----
  gemm(stream, outs, F(P_LINW), F(P_LINB), fin, 32, 512, 4608, 4608, 4608, 512, 1,
       1, 0, 0, 0, 1, 0, 0, 0, 1.f, 1);                         // ReLU
  gemm(stream, fin, F(P_CLSW), F(P_CLSB), (float*)d_out, 32, 3, 512, 512, 512, 3, 1);
}
